// EEG_GraphTransformer_44641890075107
// MI455X (gfx1250) — matrix operand order repaired and checked by reading, not compiled, not run on hardware
//
#include <hip/hip_runtime.h>
#include <hip/hip_bf16.h>
#include <math.h>

// ---------------------------------------------------------------------------
// EEG GraphTransformer for MI455X (gfx1250), wave32, WMMA bf16 w/ f32 accum.
// ---------------------------------------------------------------------------

typedef __attribute__((ext_vector_type(16))) __bf16 v16bf;
typedef __attribute__((ext_vector_type(8)))  __bf16 v8bf;
typedef __attribute__((ext_vector_type(4)))  __bf16 v4bf;
typedef __attribute__((ext_vector_type(2)))  __bf16 v2bf;
typedef __attribute__((ext_vector_type(8)))  float  v8f;
typedef __attribute__((ext_vector_type(4)))  float  v4f;

#define HID    128
#define NN     19
#define HEADS  8
#define DH     16
#define BATCH  1024
#define MROWS  (BATCH * NN)   // 19456
#define INDIM  3000
#define LN_EPS 1e-5f

// Swizzled B-matrix LDS layout: one slot of BSLOT bf16 per (ntile, lane);
// slot holds the lane's 16 fragment elements contiguously (kk = 0..15).
// BSLOT=24 -> 48B slot stride: 16B aligned, lanes land on distinct banks.
#define BSLOT 24

// ---- WMMA fragment loaders (ISA 7.12.2 wave32 layouts) --------------------
// A (16x32 bf16) from row-major LDS tile: lane L: m=L&15, hi=L>>4;
// elements: [0..7] = A[m][8hi + 0..7], [8..15] = A[m][16 + 8hi + 0..7].
__device__ __forceinline__ v16bf frag_a_lds(const __bf16* As, int lda) {
  const int lane = threadIdx.x & 31;
  const int m = lane & 15, hi = lane >> 4;
  const __bf16* r = As + m * lda + 8 * hi;
  v8bf lo = *(const v8bf*)(r);
  v8bf hi8 = *(const v8bf*)(r + 16);
  return __builtin_shufflevector(lo, hi8, 0, 1, 2, 3, 4, 5, 6, 7,
                                 8, 9, 10, 11, 12, 13, 14, 15);
}

// B fragment from swizzled layout: 16 contiguous bf16 per lane slot.
__device__ __forceinline__ v16bf frag_b_swz(const __bf16* slotBase) {
  const int lane = threadIdx.x & 31;
  const __bf16* p = slotBase + lane * BSLOT;
  v8bf lo = *(const v8bf*)(p);
  v8bf hi8 = *(const v8bf*)(p + 8);
  return __builtin_shufflevector(lo, hi8, 0, 1, 2, 3, 4, 5, 6, 7,
                                 8, 9, 10, 11, 12, 13, 14, 15);
}

__device__ __forceinline__ v8f wmma_bf16(v16bf a, v16bf b, v8f c) {
  return __builtin_amdgcn_wmma_f32_16x16x32_bf16(
      false, a, false, b, (short)0, c, false, false);
}

// Swizzled store helper: element (k in 0..31 within slice, c in 0..127),
// pair (k even, k+1) packed. slot = (ntile*32 + n + 16*((k>>3)&1)), elem = (k&7) + 8*(k>>4).
__device__ __forceinline__ void store_b_pair(__bf16* Wsw, int k, int c,
                                             float v0, float v1) {
  const int n = c & 15, nt = c >> 4;
  const int hi = (k >> 3) & 1, kk = (k & 7) + 8 * (k >> 4);
  const int slot = nt * 32 + n + 16 * hi;
  v2bf t;
  t[0] = (__bf16)v0;
  t[1] = (__bf16)v1;
  *(v2bf*)&Wsw[slot * BSLOT + kk] = t;
}

// ---------------------------------------------------------------------------
// Kernel 1: row bias = emb_h_b + pe @ emb_pe_w + emb_pe_b   (19x128)
// ---------------------------------------------------------------------------
__global__ void pe_bias_kernel(const float* __restrict__ pe,
                               const float* __restrict__ pw,
                               const float* __restrict__ pb,
                               const float* __restrict__ hb,
                               float* __restrict__ rb) {
  int idx = blockIdx.x * blockDim.x + threadIdx.x;
  if (idx >= NN * HID) return;
  int n = idx / HID, c = idx % HID;
  float s = pb[c] + hb[c];
#pragma unroll
  for (int k = 0; k < 8; ++k) s += pe[n * 8 + k] * pw[k * HID + c];
  rb[idx] = s;
}

// ---------------------------------------------------------------------------
// Kernel 2: embedding GEMM  H[m,:] = x_row(m) @ W + rowbias[m%19]
//   x is (B, K, 19): A[m][k] = x[b*K*19 + k*19 + n], m = b*19+n
//   Block: 256 thr (8 waves), 64-row M-tile, wave w owns cols [16w,16w+16).
//   93 unguarded k-steps + 1 guarded tail; per-thread base pointers hoisted.
// ---------------------------------------------------------------------------
#define EMB_LDA 40   // bf16; 80B rows -> rows land on distinct banks

__global__ void embed_kernel(const float* __restrict__ x,
                             const float* __restrict__ W,
                             const float* __restrict__ rb,
                             float* __restrict__ H, int K) {
  __shared__ __bf16 As[64 * EMB_LDA];        // 5.0 KB
  __shared__ __bf16 Wsw[8 * 32 * BSLOT];     // 12.0 KB (one 32-K slice)
  const int tid = threadIdx.x, wv = tid >> 5, lane = tid & 31;
  const int m0 = blockIdx.x * 64;

  // per-thread A base: row is constant across k-steps
  const int ar = tid & 63;
  const int akp0 = tid >> 6;                 // 0..3; pairs akp0+4p
  {
  }
  const int am = m0 + ar;
  const int ab = am / NN, an = am - ab * NN;
  const float* agp = x + (size_t)ab * K * NN + an;

  // per-thread W column (constant) and k-pair start
  const int wc = tid & 127;
  const int wkp0 = tid >> 7;                 // 0..1; pairs wkp0+2p

  v8f acc[4] = {};
  const int ktFull = K >> 5;                 // 93 full slices
  for (int kt = 0; kt < ktFull; ++kt) {
    const int kg0 = kt << 5;
    // ---- stage A: 4 k-pairs per thread, NT loads (x is streamed once) ----
    float a0[4], a1[4];
#pragma unroll
    for (int p = 0; p < 4; ++p) {
      const int kg = kg0 + 2 * (akp0 + 4 * p);
      const float* g = agp + (size_t)kg * NN;
      a0[p] = __builtin_nontemporal_load(g);
      a1[p] = __builtin_nontemporal_load(g + NN);
    }
#pragma unroll
    for (int p = 0; p < 4; ++p) {
      v2bf t;
      t[0] = (__bf16)a0[p];
      t[1] = (__bf16)a1[p];
      *(v2bf*)&As[ar * EMB_LDA + 2 * (akp0 + 4 * p)] = t;
    }
    // ---- stage W slice swizzled: 8 k-pairs per thread ----
    float w0[8], w1[8];
#pragma unroll
    for (int p = 0; p < 8; ++p) {
      const int kp = wkp0 + 2 * p;           // 0..15
      const float* g = W + (size_t)(kg0 + 2 * kp) * HID + wc;
      w0[p] = g[0];
      w1[p] = g[HID];
    }
#pragma unroll
    for (int p = 0; p < 8; ++p) {
      const int kp = wkp0 + 2 * p;
      store_b_pair(Wsw, 2 * kp, wc, w0[p], w1[p]);
    }
    __syncthreads();
    const v16bf bfr = frag_b_swz(&Wsw[(wv * 32) * BSLOT]);
#pragma unroll
    for (int mt = 0; mt < 4; ++mt) {
      v16bf afr = frag_a_lds(&As[mt * 16 * EMB_LDA], EMB_LDA);
      acc[mt] = wmma_bf16(afr, bfr, acc[mt]);
    }
    __syncthreads();
  }
  // ---- guarded tail (K % 32 leftover, zero-padded) ----
  if (K & 31) {
    const int kg0 = ktFull << 5;
#pragma unroll
    for (int p = 0; p < 4; ++p) {
      const int kg = kg0 + 2 * (akp0 + 4 * p);
      v2bf t;
      t[0] = (__bf16)((kg < K) ? agp[(size_t)kg * NN] : 0.f);
      t[1] = (__bf16)((kg + 1 < K) ? agp[(size_t)(kg + 1) * NN] : 0.f);
      *(v2bf*)&As[ar * EMB_LDA + 2 * (akp0 + 4 * p)] = t;
    }
#pragma unroll
    for (int p = 0; p < 8; ++p) {
      const int kp = wkp0 + 2 * p;
      const int kg = kg0 + 2 * kp;
      float v0 = (kg < K) ? W[(size_t)kg * HID + wc] : 0.f;
      float v1 = (kg + 1 < K) ? W[(size_t)(kg + 1) * HID + wc] : 0.f;
      store_b_pair(Wsw, 2 * kp, wc, v0, v1);
    }
    __syncthreads();
    const v16bf bfr = frag_b_swz(&Wsw[(wv * 32) * BSLOT]);
#pragma unroll
    for (int mt = 0; mt < 4; ++mt) {
      v16bf afr = frag_a_lds(&As[mt * 16 * EMB_LDA], EMB_LDA);
      acc[mt] = wmma_bf16(afr, bfr, acc[mt]);
    }
  }
  // epilogue: C/D layout -> row r+8*hi, col = 16*wv + (lane&15)
  const int n = lane & 15, hi = lane >> 4;
  const int col = wv * 16 + n;
#pragma unroll
  for (int mt = 0; mt < 4; ++mt) {
#pragma unroll
    for (int r = 0; r < 8; ++r) {
      int m = m0 + mt * 16 + r + 8 * hi;
      int node = m % NN;
      H[(size_t)m * HID + col] = acc[mt][r] + rb[node * HID + col];
    }
  }
}

// ---------------------------------------------------------------------------
// Kernel 3: generic  C = act(A @ W + bias [+ res]),  A:(M,128) W:(128,128)
//   W staged swizzled once per block (B-fragments then live in registers);
//   grid-stride over 64-row tiles; A staged via float4 -> v4bf.
// ---------------------------------------------------------------------------
#define G_LDA 136

__global__ void gemm128_kernel(const float* __restrict__ A,
                               const float* __restrict__ W,
                               const float* __restrict__ bias,
                               const float* __restrict__ res,
                               float* __restrict__ C, int M, int act) {
  __shared__ __bf16 Wsw[4 * 8 * 32 * BSLOT];  // 48.0 KB (4 k-slices)
  __shared__ __bf16 As[64 * G_LDA];           // 17.0 KB
  const int tid = threadIdx.x, wv = tid >> 5, lane = tid & 31;
  const int wc = tid & 127;
  // stage whole W (64 k-pairs x 128 c = 8192 pairs; 32 per thread)
#pragma unroll
  for (int p = 0; p < 32; ++p) {
    const int pairIdx = tid + p * 256;
    const int kp = pairIdx >> 7;              // 0..63
    const int k = 2 * kp;
    const float* g = W + (size_t)k * HID + wc;
    const float v0 = g[0], v1 = g[HID];
    const int ks = k >> 5, kin = k & 31;
    store_b_pair(&Wsw[(ks * 8 * 32) * BSLOT], kin, wc, v0, v1);
  }
  __syncthreads();
  v16bf bfr[4];
#pragma unroll
  for (int ks = 0; ks < 4; ++ks)
    bfr[ks] = frag_b_swz(&Wsw[((ks * 8 + wv) * 32) * BSLOT]);

  const int tiles = (M + 63) >> 6;
  for (int t = blockIdx.x; t < tiles; t += gridDim.x) {
    const int m0 = t * 64;
    __syncthreads();  // As reuse across grid-stride iterations
#pragma unroll
    for (int p = 0; p < 8; ++p) {
      const int idx = tid + p * 256;          // 0..2047 float4 slots
      const int r = idx >> 5;                 // 0..63
      const int cv = (idx & 31) * 4;
      const int m = m0 + r;
      v4f v = {0.f, 0.f, 0.f, 0.f};
      if (m < M) v = *(const v4f*)(A + (size_t)m * HID + cv);
      v4bf t;
      t[0] = (__bf16)v.x;
      t[1] = (__bf16)v.y;
      t[2] = (__bf16)v.z;
      t[3] = (__bf16)v.w;
      *(v4bf*)&As[r * G_LDA + cv] = t;
    }
    __syncthreads();
    v8f acc[4] = {};
#pragma unroll
    for (int mt = 0; mt < 4; ++mt) {
#pragma unroll
      for (int ks = 0; ks < 4; ++ks) {
        v16bf afr = frag_a_lds(&As[(mt * 16) * G_LDA + ks * 32], G_LDA);
        acc[mt] = wmma_bf16(afr, bfr[ks], acc[mt]);
      }
    }
    const int n = lane & 15, hi = lane >> 4;
    const int col = wv * 16 + n;
    const float bv = bias ? bias[col] : 0.f;
#pragma unroll
    for (int mt = 0; mt < 4; ++mt) {
#pragma unroll
      for (int r = 0; r < 8; ++r) {
        int m = m0 + mt * 16 + r + 8 * hi;
        if (m < M) {
          float v = acc[mt][r] + bv;
          if (res) v += res[(size_t)m * HID + col];
          if (act) v = fmaxf(v, 0.f);
          C[(size_t)m * HID + col] = v;
        }
      }
    }
  }
}

// ---------------------------------------------------------------------------
// Kernel 4: attention. Complete graph minus diagonal; softmax over ALL 342
// edges per (b,h) (reference softmax axis=1). Block = batch, wave = head.
// ---------------------------------------------------------------------------
__global__ void attn_kernel(const float* __restrict__ Q,
                            const float* __restrict__ Km,
                            const float* __restrict__ V,
                            float* __restrict__ AGG) {
  __shared__ float Qs[NN * HID];
  __shared__ float Ks[NN * HID];
  __shared__ float Vs[NN * HID];
  __shared__ float S[HEADS][NN * NN + 1];
  const int b = blockIdx.x;
  const int tid = threadIdx.x, h = tid >> 5, lane = tid & 31;
  const size_t base = (size_t)b * NN * HID;
  for (int i = tid; i < NN * HID; i += 256) {
    Qs[i] = Q[base + i];
    Ks[i] = Km[base + i];
    Vs[i] = V[base + i];
  }
  __syncthreads();
  const float scale = 0.25f;  // 1/sqrt(DH=16)
  float psum = 0.f;
  for (int e = lane; e < NN * NN; e += 32) {
    int i = e / NN, j = e - i * NN;  // src=i, dst=j
    float es = 0.f;
    if (i != j) {
      const float* qp = &Qs[j * HID + h * DH];
      const float* kp = &Ks[i * HID + h * DH];
      float s = 0.f;
#pragma unroll
      for (int d = 0; d < DH; ++d) s += qp[d] * kp[d];
      s *= scale;
      s = fminf(5.f, fmaxf(-5.f, s));
      es = __expf(s);
    }
    S[h][e] = es;
    psum += es;
  }
#pragma unroll
  for (int off = 16; off > 0; off >>= 1) psum += __shfl_xor(psum, off, 32);
  const float inv = 1.f / psum;
  for (int idx = lane; idx < NN * DH; idx += 32) {
    int j = idx / DH, d = idx - j * DH;
    float a = 0.f;
#pragma unroll
    for (int i = 0; i < NN; ++i) a += S[h][i * NN + j] * Vs[i * HID + h * DH + d];
    AGG[base + j * HID + h * DH + d] = a * inv;
  }
}

// ---------------------------------------------------------------------------
// Kernel 5: LayerNorm in place, one wave per 128-wide row.
// ---------------------------------------------------------------------------
__global__ void ln_kernel(float* __restrict__ X, const float* __restrict__ g,
                          const float* __restrict__ bta, int M) {
  const int tid = threadIdx.x, wv = tid >> 5, lane = tid & 31;
  const int m = blockIdx.x * 8 + wv;
  if (m >= M) return;
  float* row = X + (size_t)m * HID;
  float x[4];
  float s = 0.f;
#pragma unroll
  for (int p = 0; p < 4; ++p) { x[p] = row[lane + 32 * p]; s += x[p]; }
#pragma unroll
  for (int off = 16; off > 0; off >>= 1) s += __shfl_xor(s, off, 32);
  const float mean = s * (1.f / 128.f);
  float vs = 0.f;
#pragma unroll
  for (int p = 0; p < 4; ++p) { float d = x[p] - mean; vs += d * d; }
#pragma unroll
  for (int off = 16; off > 0; off >>= 1) vs += __shfl_xor(vs, off, 32);
  const float rstd = rsqrtf(vs * (1.f / 128.f) + LN_EPS);
#pragma unroll
  for (int p = 0; p < 4; ++p) {
    int c = lane + 32 * p;
    row[c] = (x[p] - mean) * rstd * g[c] + bta[c];
  }
}

// ---------------------------------------------------------------------------
// Kernel 6: mean-pool over 19 nodes -> (1024,128)
// ---------------------------------------------------------------------------
__global__ void pool_kernel(const float* __restrict__ H, float* __restrict__ P) {
  int idx = blockIdx.x * blockDim.x + threadIdx.x;
  if (idx >= BATCH * HID) return;
  int b = idx >> 7, c = idx & 127;
  float s = 0.f;
#pragma unroll
  for (int n = 0; n < NN; ++n) s += H[((size_t)b * NN + n) * HID + c];
  P[idx] = s * (1.f / 19.f);
}

// ---------------------------------------------------------------------------
// Kernel 7: final 128->4 projection
// ---------------------------------------------------------------------------
__global__ void out_kernel(const float* __restrict__ Z, const float* __restrict__ W2,
                           const float* __restrict__ b2, float* __restrict__ out) {
  int idx = blockIdx.x * blockDim.x + threadIdx.x;
  if (idx >= BATCH * 4) return;
  int b = idx >> 2, o = idx & 3;
  float s = b2[o];
#pragma unroll 8
  for (int k = 0; k < HID; ++k) s += Z[b * HID + k] * W2[k * 4 + o];
  out[idx] = s;
}

// ---------------------------------------------------------------------------
extern "C" void kernel_launch(void* const* d_in, const int* in_sizes, int n_in,
                              void* d_out, int out_size, void* d_ws, size_t ws_size,
                              hipStream_t stream) {
  (void)in_sizes; (void)n_in; (void)out_size; (void)ws_size;
  const float* node_features = (const float*)d_in[0];
  const float* pe       = (const float*)d_in[1];
  // d_in[2] edge_index: complete graph minus diagonal (fixed), handled densely.
  const float* emb_h_w  = (const float*)d_in[3];
  const float* emb_h_b  = (const float*)d_in[4];
  const float* emb_pe_w = (const float*)d_in[5];
  const float* emb_pe_b = (const float*)d_in[6];
  const float* wq_w = (const float*)d_in[7];
  const float* wq_b = (const float*)d_in[8];
  const float* wk_w = (const float*)d_in[9];
  const float* wk_b = (const float*)d_in[10];
  const float* wv_w = (const float*)d_in[11];
  const float* wv_b = (const float*)d_in[12];
  const float* wo_w = (const float*)d_in[13];
  const float* wo_b = (const float*)d_in[14];
  const float* ln1_g = (const float*)d_in[15];
  const float* ln1_b = (const float*)d_in[16];
  const float* lin1_w = (const float*)d_in[17];
  const float* lin1_b = (const float*)d_in[18];
  const float* lin2_w = (const float*)d_in[19];
  const float* lin2_b = (const float*)d_in[20];
  const float* ln2_g = (const float*)d_in[21];
  const float* ln2_b = (const float*)d_in[22];
  const float* mlp_w0 = (const float*)d_in[23];
  const float* mlp_b0 = (const float*)d_in[24];
  const float* mlp_w1 = (const float*)d_in[25];
  const float* mlp_b1 = (const float*)d_in[26];
  const float* mlp_w2 = (const float*)d_in[27];
  const float* mlp_b2 = (const float*)d_in[28];

  float* ws = (float*)d_ws;
  const size_t S = (size_t)MROWS * HID;
  float* H  = ws;
  float* Qb = ws + 1 * S;
  float* Kb = ws + 2 * S;
  float* Vb = ws + 3 * S;
  float* AG = ws + 4 * S;
  float* FF = ws + 5 * S;
  float* RB = ws + 6 * S;
  float* P  = RB + NN * HID;
  float* Z1 = P + BATCH * HID;
  float* Z2 = Z1 + BATCH * HID;

  pe_bias_kernel<<<(NN * HID + 255) / 256, 256, 0, stream>>>(pe, emb_pe_w, emb_pe_b, emb_h_b, RB);
  embed_kernel<<<MROWS / 64, 256, 0, stream>>>(node_features, emb_h_w, RB, H, INDIM);

  for (int l = 0; l < 3; ++l) {
    const int wo_off = l * HID * HID;
    const int bo_off = l * HID;
    gemm128_kernel<<<304, 256, 0, stream>>>(H, wq_w + wo_off, wq_b + bo_off, nullptr, Qb, MROWS, 0);
    gemm128_kernel<<<304, 256, 0, stream>>>(H, wk_w + wo_off, wk_b + bo_off, nullptr, Kb, MROWS, 0);
    gemm128_kernel<<<304, 256, 0, stream>>>(H, wv_w + wo_off, wv_b + bo_off, nullptr, Vb, MROWS, 0);
    attn_kernel<<<BATCH, 256, 0, stream>>>(Qb, Kb, Vb, AG);
    gemm128_kernel<<<304, 256, 0, stream>>>(AG, wo_w + wo_off, wo_b + bo_off, H, H, MROWS, 0);
    ln_kernel<<<MROWS / 8, 256, 0, stream>>>(H, ln1_g + bo_off, ln1_b + bo_off, MROWS);
    gemm128_kernel<<<304, 256, 0, stream>>>(H, lin1_w + wo_off, lin1_b + bo_off, nullptr, FF, MROWS, 1);
    gemm128_kernel<<<304, 256, 0, stream>>>(FF, lin2_w + wo_off, lin2_b + bo_off, H, H, MROWS, 0);
    ln_kernel<<<MROWS / 8, 256, 0, stream>>>(H, ln2_g + bo_off, ln2_b + bo_off, MROWS);
  }

  pool_kernel<<<(BATCH * HID + 255) / 256, 256, 0, stream>>>(H, P);
  gemm128_kernel<<<16, 256, 0, stream>>>(P, mlp_w0, mlp_b0, nullptr, Z1, BATCH, 1);
  gemm128_kernel<<<16, 256, 0, stream>>>(Z1, mlp_w1, mlp_b1, nullptr, Z2, BATCH, 1);
  out_kernel<<<(BATCH * 4 + 255) / 256, 256, 0, stream>>>(Z2, mlp_w2, mlp_b2, (float*)d_out);
}
